// RangeGRU_85993835200591
// MI455X (gfx1250) — matrix operand order repaired and checked by reading, not compiled, not run on hardware
//
#include <hip/hip_runtime.h>

#define SEQ   2048
#define BATCH 64
#define ISZ   512
#define HSZ   512
#define GSZ   1024   // only z and n gate columns (r is dead in the reference)

typedef __attribute__((ext_vector_type(16))) __bf16       v16bf;
typedef __attribute__((ext_vector_type(8)))  __bf16       v8bf;
typedef __attribute__((ext_vector_type(8)))  float        v8f;
typedef __attribute__((ext_vector_type(4)))  unsigned int v4u;

union Frag16 { struct { v4u lo; v4u hi; } u; v16bf v; };

// B fragment (32x16 K x N): lane n<16 holds col n K=0..15, lanes 16-31 K=16..31.
// Source row-major [col][K] => 16 contiguous bf16 per lane (two b128 loads).
__device__ __forceinline__ v16bf load_b_frag(const __bf16* __restrict__ w, int ldk,
                                             int colBase, int kBase, int lane) {
  const __bf16* p = w + (size_t)(colBase + (lane & 15)) * ldk + kBase + ((lane >> 4) << 4);
  Frag16 f;
  f.u.lo = *(const v4u*)(p);
  f.u.hi = *(const v4u*)(p + 8);
  return f.v;
}

// A fragment (16x32 M x K): lane L row=L&15, elems 0..15 = K b..b+15,
// b = (L>>4)*16. Works for global bf16 x and LDS bf16 h (row-major, ld = K size).
__device__ __forceinline__ v16bf load_a_frag(const __bf16* __restrict__ m, int ldk,
                                             int kBase, int lane) {
  const __bf16* p = m + (size_t)(lane & 15) * ldk + kBase + ((lane >> 4) << 4);
  Frag16 f;
  f.u.lo = *(const v4u*)(p);
  f.u.hi = *(const v4u*)(p + 8);
  return f.v;
}

__device__ __forceinline__ v8f wmma_bf16(v16bf a, v16bf b, v8f c) {
  return __builtin_amdgcn_wmma_f32_16x16x32_bf16(false, a, false, b, (short)0, c,
                                                 false, false);
}

// ---- Kernel 1: convert live gate rows of W_ih / W_hh to bf16 (vectorized) ---
__global__ __launch_bounds__(256) void cvt_weights(
    const float* __restrict__ wih, const float* __restrict__ whh,
    __bf16* __restrict__ wih_bf, __bf16* __restrict__ whh_bf) {
  const int i = (blockIdx.x * 256 + threadIdx.x) * 8;
  if (i < GSZ * ISZ) {
    v8f a = *(const v8f*)(wih + (size_t)HSZ * ISZ + i);  // skip dead r-gate rows
    v8f b = *(const v8f*)(whh + (size_t)HSZ * HSZ + i);
    *(v8bf*)(wih_bf + i) = __builtin_convertvector(a, v8bf);
    *(v8bf*)(whh_bf + i) = __builtin_convertvector(b, v8bf);
  }
}

// ---- Kernel 2: convert x to bf16 once (removes cvt from GEMM inner loop) ----
__global__ __launch_bounds__(256) void cvt_x(const float* __restrict__ x,
                                             __bf16* __restrict__ xb) {
  const size_t i = ((size_t)blockIdx.x * 256 + threadIdx.x) * 8;
  v8f f = *(const v8f*)(x + i);
  *(v8bf*)(xb + i) = __builtin_convertvector(f, v8bf);
}

// ---- Kernel 3: x_proj[131072,1024] = x_bf @ W_ih_zn^T + bias ----------------
// One wave: 16(M) x 64(N), software-pipelined one k-step ahead. Stalls per wave
// are covered by high occupancy (131072 waves, ~64 VGPRs); kernel is DRAM-bound.
__global__ __launch_bounds__(256) void xproj_gemm(
    const __bf16* __restrict__ xb, const __bf16* __restrict__ wih_bf,
    const float* __restrict__ bias, float* __restrict__ xproj) {
  const int lane   = threadIdx.x & 31;
  const int wave   = blockIdx.x * 8 + (threadIdx.x >> 5);
  const int mTile  = wave >> 4;          // 8192 M-tiles
  const int nGroup = wave & 15;          // 16 groups of 64 cols
  const size_t rowBase = (size_t)mTile * 16;
  const int colBase = nGroup * 64;
  const __bf16* xrow = xb + rowBase * ISZ;

  v8f acc[4] = {};
  v16bf a = load_a_frag(xrow, ISZ, 0, lane);
  v16bf b[4];
#pragma unroll
  for (int t = 0; t < 4; ++t) b[t] = load_b_frag(wih_bf, ISZ, colBase + t * 16, 0, lane);

#pragma unroll
  for (int k = 0; k < ISZ; k += 32) {
    const int kn = (k + 32) & (ISZ - 1);          // wraps harmlessly on last iter
    v16bf an = load_a_frag(xrow, ISZ, kn, lane);
    v16bf bn[4];
#pragma unroll
    for (int t = 0; t < 4; ++t) bn[t] = load_b_frag(wih_bf, ISZ, colBase + t * 16, kn, lane);
#pragma unroll
    for (int t = 0; t < 4; ++t) acc[t] = wmma_bf16(a, b[t], acc[t]);
    a = an;
#pragma unroll
    for (int t = 0; t < 4; ++t) b[t] = bn[t];
  }

  // C/D layout: (lane, reg r) -> row (lane>>4)*8 + r, col lane&15
  const int n    = lane & 15;
  const int mOff = (lane >> 4) << 3;
#pragma unroll
  for (int t = 0; t < 4; ++t) {
    const int col = colBase + t * 16 + n;
    const float bv = bias[HSZ + col];
#pragma unroll
    for (int r = 0; r < 8; ++r)
      xproj[(rowBase + mOff + r) * GSZ + col] = acc[t][r] + bv;
  }
}

// ---- Kernel 4: sequential scan, 4 independent batch-group blocks ------------
// Wave w owns the z-tile (gate cols 16w..16w+16) AND the matching n-tile
// (cols 512+16w..): z and n for h-column j land in the same thread's C slots,
// so the GRU update is register-resident. h ping-pongs between two LDS buffers
// -> one barrier per timestep, no gates exchange through LDS.
__global__ __launch_bounds__(1024) void gru_scan(
    const float* __restrict__ xproj, const __bf16* __restrict__ whh_bf,
    float* __restrict__ out, float* __restrict__ hfin) {
  extern __shared__ char smem[];
  __bf16* hping = (__bf16*)smem;                     // [16][512] 16 KB
  __bf16* hpong = (__bf16*)(smem + 16 * HSZ * 2);    // [16][512] 16 KB

  const int tid   = threadIdx.x;
  const int lane  = tid & 31;
  const int wave  = tid >> 5;        // 0..31 -> h columns [16w, 16w+16)
  const int zCol  = wave * 16;       // z gate tile base (local gate col space)
  const int nCol  = HSZ + zCol;      // n gate tile base
  const int bRow0 = blockIdx.x * 16;

  for (int i = tid; i < 16 * HSZ; i += 1024) {
    hping[i] = (__bf16)0.0f;
    hpong[i] = (__bf16)0.0f;
  }
  const int n    = lane & 15;
  const int mOff = (lane >> 4) << 3;
  const int jcol = zCol + n;         // h column covered by this thread's C slots
  float hreg[8];
#pragma unroll
  for (int r = 0; r < 8; ++r) hreg[r] = 0.f;
  __syncthreads();

  __bf16* hcur = hping;
  __bf16* hnxt = hpong;
  for (int t = 0; t < SEQ; ++t) {
    // accz/accn = h_bf16 @ W_hh_zn^T for this wave's z/n tiles (pipelined)
    v8f accz = {}, accn = {};
    v16bf a   = load_a_frag(hcur, HSZ, 0, lane);
    v16bf bz  = load_b_frag(whh_bf, HSZ, zCol, 0, lane);
    v16bf bnn = load_b_frag(whh_bf, HSZ, nCol, 0, lane);
#pragma unroll
    for (int k = 0; k < HSZ; k += 32) {
      const int kn = (k + 32) & (HSZ - 1);
      v16bf a2   = load_a_frag(hcur, HSZ, kn, lane);
      v16bf bz2  = load_b_frag(whh_bf, HSZ, zCol, kn, lane);
      v16bf bnn2 = load_b_frag(whh_bf, HSZ, nCol, kn, lane);
      accz = wmma_bf16(a, bz, accz);
      accn = wmma_bf16(a, bnn, accn);
      a = a2; bz = bz2; bnn = bnn2;
    }
    if (t + 1 < SEQ)   // warm next step's 64 KB x_proj slab (global_prefetch_b8)
      __builtin_prefetch(xproj + ((size_t)(t + 1) * BATCH + bRow0) * GSZ + tid * 16, 0, 0);

    // register-resident GRU update on this thread's 8 (row, jcol) slots
    const float* xp = xproj + ((size_t)t * BATCH + bRow0) * GSZ;
#pragma unroll
    for (int r = 0; r < 8; ++r) {
      const int row = mOff + r;
      const float z  = 1.f / (1.f + __expf(-(accz[r] + xp[row * GSZ + zCol + n])));
      const float nn = tanhf(accn[r] + xp[row * GSZ + nCol + n]);
      const float hn = (1.f - z) * nn + z * hreg[r];
      hreg[r] = hn;
      hnxt[row * HSZ + jcol] = (__bf16)hn;
      out[((size_t)t * BATCH + bRow0 + row) * HSZ + jcol] = hn;
    }
    __syncthreads();                 // writes to hnxt + reads of hcur both done
    __bf16* tmp = hcur; hcur = hnxt; hnxt = tmp;
  }
#pragma unroll
  for (int r = 0; r < 8; ++r)
    hfin[(size_t)(bRow0 + mOff + r) * HSZ + jcol] = hreg[r];
}

extern "C" void kernel_launch(void* const* d_in, const int* in_sizes, int n_in,
                              void* d_out, int out_size, void* d_ws, size_t ws_size,
                              hipStream_t stream) {
  const float* x    = (const float*)d_in[0];
  const float* wih  = (const float*)d_in[1];
  const float* whh  = (const float*)d_in[2];
  const float* bias = (const float*)d_in[3];
  float* out = (float*)d_out;

  char* ws = (char*)d_ws;
  const size_t W_BYTES  = (size_t)GSZ * ISZ * 2;             // 1 MB each
  const size_t XB_BYTES = (size_t)SEQ * BATCH * ISZ * 2;     // 128 MB
  __bf16* wih_bf = (__bf16*)ws;
  __bf16* whh_bf = (__bf16*)(ws + W_BYTES);
  __bf16* xb     = (__bf16*)(ws + 2 * W_BYTES);
  float*  xproj  = (float*)(ws + 2 * W_BYTES + XB_BYTES);    // 512 MB

  cvt_weights<<<(GSZ * ISZ / 8 + 255) / 256, 256, 0, stream>>>(wih, whh, wih_bf, whh_bf);
  cvt_x<<<(int)((size_t)SEQ * BATCH * ISZ / 8 / 256), 256, 0, stream>>>(x, xb);

  // 8192 M-tiles * 16 N-groups = 131072 waves, 8 waves/block
  xproj_gemm<<<16384, 256, 0, stream>>>(xb, wih_bf, bias, xproj);

  const size_t smem = (size_t)2 * 16 * HSZ * 2;  // 32 KB ping-pong h buffers
  gru_scan<<<BATCH / 16, 1024, smem, stream>>>(xproj, whh_bf, out,
                                               out + (size_t)SEQ * BATCH * HSZ);
}
